// DecoderLayer_44195213476316
// MI455X (gfx1250) — matrix operand order repaired and checked by reading, not compiled, not run on hardware
//
#include <hip/hip_runtime.h>
#include <math.h>
#include <stdint.h>

#define S_  2048
#define B_  2
#define D_  1024
#define H_  16
#define DH_ 64
#define FF_ 4096
#define M_  (B_*S_)   // 4096 tokens

typedef __attribute__((ext_vector_type(16))) _Float16 v16h;
typedef __attribute__((ext_vector_type(8)))  float    v8f;

union Frag { v16h v; uint4 q[2]; };

// A-matrix 16x32 f16 fragment (ISA 7.12.2): lane<16 holds K {k0..k0+7, k0+16..k0+23},
// lane>=16 holds K {k0+8..k0+15, k0+24..k0+31}; rowp = per-lane row pointer.
__device__ __forceinline__ v16h load_af(const _Float16* rowp, int lane, int k0){
  Frag f;
  int ka = (lane < 16) ? k0 : (k0 + 8);
  f.q[0] = *(const uint4*)(rowp + ka);
  f.q[1] = *(const uint4*)(rowp + ka + 16);
  return f.v;
}
// B-matrix 32x16 f16 fragment: lane<16 -> K {k0..k0+7, k0+16..k0+23}, lane>=16 -> K {k0+8..k0+15, k0+24..k0+31},
// column (N) fixed per lane; rowp points at that column's contiguous-K data.
__device__ __forceinline__ v16h load_bf(const _Float16* rowp, int lane, int k0){
  Frag f;
  int kb = (lane < 16) ? k0 : (k0 + 8);
  f.q[0] = *(const uint4*)(rowp + kb);
  f.q[1] = *(const uint4*)(rowp + kb + 16);
  return f.v;
}
__device__ __forceinline__ v8f wmma16(v16h a, v16h b, v8f c){
  return __builtin_amdgcn_wmma_f32_16x16x32_f16(false, a, false, b, (short)0, c, false, false);
}
__device__ __forceinline__ float gelu_f(float x){
  return 0.5f * x * (1.0f + erff(x * 0.70710678118654752f));
}

// ---------------- GEMM: C[M,N] = A[M,K](f16) * WT[N,K](f16) + bias, opt GELU/residual ----
// Double-buffered LDS; next Wt tile staged via async global->LDS DMA (ASYNCcnt).
template<int N, int K, bool GELU, bool RESID, bool OUTF16>
__global__ __launch_bounds__(256) void gemm_k(const _Float16* __restrict__ A,
                                              const _Float16* __restrict__ WT,
                                              const float* __restrict__ bias,
                                              const float* __restrict__ R,
                                              void* __restrict__ Cout){
  __shared__ __align__(16) _Float16 sB[2][128*40];  // 2 x (128 N-rows x 32 K, stride 40)
  const int lane = threadIdx.x & 31, wave = threadIdx.x >> 5;
  const int m0 = blockIdx.x*128 + wave*16;
  const int n0 = blockIdx.y*128;
  const _Float16* arow = A + (size_t)(m0 + (lane & 15)) * K;

  // Cooperative async stage of Wt[n0..n0+127][k0..k0+31] into sB[buf].
  // 512 b128 granules / 256 threads = 2 per thread, fully unrolled (no EXEC games).
  auto stage_async = [&](int k0, int buf){
    int g = threadIdx.x;
    #pragma unroll
    for(int it = 0; it < 2; ++it, g += 256){
      int r = g >> 2, c = g & 3;
      uint32_t lds = (uint32_t)(uintptr_t)(&sB[buf][0] + r*40 + c*8);   // flat->LDS: low 32 bits
      uint64_t ga  = (uint64_t)(uintptr_t)(WT + (size_t)(n0 + r)*K + k0 + c*8);
      asm volatile("global_load_async_to_lds_b128 %0, %1, off"
                   :: "v"(lds), "v"(ga) : "memory");
    }
  };

  v8f acc[8] = {};
  stage_async(0, 0);
  asm volatile("s_wait_asynccnt 0" ::: "memory");
  __syncthreads();
  int buf = 0;
  for(int k0 = 0; k0 < K; k0 += 32){
    if (k0 + 32 < K) stage_async(k0 + 32, buf ^ 1);   // overlap DMA with WMMA burst
    v16h a = load_af(arow, lane, k0);
    v16h bfr[8];
    #pragma unroll
    for(int t = 0; t < 8; t++)
      bfr[t] = load_bf(&sB[buf][0] + (size_t)(t*16 + (lane & 15))*40, lane, 0);
    #pragma unroll
    for(int t = 0; t < 8; t++)
      acc[t] = wmma16(a, bfr[t], acc[t]);
    asm volatile("s_wait_asynccnt 0" ::: "memory");
    __syncthreads();
    buf ^= 1;
  }
  const int colh = lane & 15, rb = (lane < 16) ? 0 : 8;   // C/D layout: vgpr r -> row r(+8)
  #pragma unroll
  for(int t = 0; t < 8; t++){
    int n = n0 + t*16 + colh;
    float bz = bias ? bias[n] : 0.0f;
    #pragma unroll
    for(int r = 0; r < 8; r++){
      size_t idx = (size_t)(m0 + rb + r)*N + n;
      float v = acc[t][r] + bz;
      if (GELU)  v = gelu_f(v);
      if (RESID) v += R[idx];
      if (OUTF16) ((_Float16*)Cout)[idx] = (_Float16)v;
      else        ((float*)Cout)[idx]    = v;
    }
  }
}

// ---------------- Flash attention: per-wave 16 queries, streaming softmax ----------------
// Q,K,O: [B,S,D] f16 (head offset h*64); VT: [B,H,64,S] f16 (pre-transposed V).
template<bool CAUSAL>
__global__ __launch_bounds__(256) void flash_k(const _Float16* __restrict__ Qp,
                                               const _Float16* __restrict__ Kp,
                                               const _Float16* __restrict__ VT,
                                               _Float16* __restrict__ Op){
  __shared__ __align__(16) _Float16 sP[8][16*40];   // per-wave P tile (16x32, stride 40)
  const int lane = threadIdx.x & 31, wave = threadIdx.x >> 5;
  const int b = blockIdx.z, h = blockIdx.y;
  const int q0 = blockIdx.x*128 + wave*16;
  const int colh = lane & 15, rb = (lane < 16) ? 0 : 8;
  const _Float16* qrow = Qp + (size_t)(b*S_ + q0 + colh)*D_ + h*DH_;
  v16h qa0 = load_af(qrow, lane, 0);
  v16h qa1 = load_af(qrow, lane, 32);
  v8f o0 = {}, o1 = {}, o2 = {}, o3 = {};
  float mrow[8], lrow[8];
  #pragma unroll
  for(int r = 0; r < 8; r++){ mrow[r] = -1e30f; lrow[r] = 0.0f; }
  int kend = CAUSAL ? ((q0 + 16 + 31) & ~31) : S_;
  if (kend > S_) kend = S_;
  _Float16* pw = &sP[wave][0];
  const _Float16* vbase = VT + ((size_t)(b*H_ + h)*DH_ + colh)*S_;
  for(int kk = 0; kk < kend; kk += 32){
    const _Float16* krow0 = Kp + (size_t)(b*S_ + kk + colh)*D_ + h*DH_;
    const _Float16* krow1 = krow0 + (size_t)16*D_;
    v8f s0 = {}, s1 = {};
    s0 = wmma16(qa0, load_bf(krow0, lane, 0),  s0);
    s0 = wmma16(qa1, load_bf(krow0, lane, 32), s0);
    s1 = wmma16(qa0, load_bf(krow1, lane, 0),  s1);
    s1 = wmma16(qa1, load_bf(krow1, lane, 32), s1);
    // Issue V fragment loads early: latency hides under the softmax VALU work below.
    const _Float16* vr = vbase + kk;
    v16h vb0 = load_bf(vr,                 lane, 0);
    v16h vb1 = load_bf(vr + (size_t)16*S_, lane, 0);
    v16h vb2 = load_bf(vr + (size_t)32*S_, lane, 0);
    v16h vb3 = load_bf(vr + (size_t)48*S_, lane, 0);
    #pragma unroll
    for(int r = 0; r < 8; r++){
      int q = q0 + rb + r;
      float a = s0[r]*0.125f;                 // 1/sqrt(64)
      float c = s1[r]*0.125f;
      if (CAUSAL){
        if (kk + colh > q)       a = -1e30f;
        if (kk + 16 + colh > q)  c = -1e30f;
      }
      float mx = fmaxf(a, c);
      #pragma unroll
      for(int off = 1; off < 16; off <<= 1) mx = fmaxf(mx, __shfl_xor(mx, off, 32));
      float mnew  = fmaxf(mrow[r], mx);
      float alpha = __expf(mrow[r] - mnew);
      float p0 = __expf(a - mnew), p1 = __expf(c - mnew);
      float ps = p0 + p1;
      #pragma unroll
      for(int off = 1; off < 16; off <<= 1) ps += __shfl_xor(ps, off, 32);
      lrow[r] = lrow[r]*alpha + ps;
      mrow[r] = mnew;
      o0[r] *= alpha; o1[r] *= alpha; o2[r] *= alpha; o3[r] *= alpha;
      pw[(rb + r)*40 + colh]      = (_Float16)p0;   // P in C-layout -> LDS (row-major 16x32)
      pw[(rb + r)*40 + 16 + colh] = (_Float16)p1;
    }
    asm volatile("s_wait_dscnt 0" ::: "memory");    // wave-local LDS transpose handoff
    v16h pa = load_af(pw + colh*40, lane, 0);       // reload in A-layout
    o0 = wmma16(pa, vb0, o0);
    o1 = wmma16(pa, vb1, o1);
    o2 = wmma16(pa, vb2, o2);
    o3 = wmma16(pa, vb3, o3);
  }
  #pragma unroll
  for(int r = 0; r < 8; r++){
    float inv = 1.0f / lrow[r];
    size_t base = (size_t)(b*S_ + q0 + rb + r)*D_ + h*DH_ + colh;
    Op[base]      = (_Float16)(o0[r]*inv);
    Op[base + 16] = (_Float16)(o1[r]*inv);
    Op[base + 32] = (_Float16)(o2[r]*inv);
    Op[base + 48] = (_Float16)(o3[r]*inv);
  }
}

// ---------------- LayerNorm (f32 in -> f16 out) ----------------
__global__ __launch_bounds__(256) void ln_k(const float* __restrict__ x,
                                            const float* __restrict__ g,
                                            const float* __restrict__ be,
                                            _Float16* __restrict__ y){
  __shared__ float s1[256], s2[256];
  const float* xr = x + (size_t)blockIdx.x*D_;
  float a = 0.f, bb = 0.f;
  for(int i = threadIdx.x; i < D_; i += 256){ float v = xr[i]; a += v; bb += v*v; }
  s1[threadIdx.x] = a; s2[threadIdx.x] = bb;
  __syncthreads();
  for(int o = 128; o > 0; o >>= 1){
    if (threadIdx.x < o){ s1[threadIdx.x] += s1[threadIdx.x+o]; s2[threadIdx.x] += s2[threadIdx.x+o]; }
    __syncthreads();
  }
  float mu  = s1[0]*(1.0f/D_);
  float var = s2[0]*(1.0f/D_) - mu*mu;
  float rs  = rsqrtf(var + 1e-5f);
  _Float16* yr = y + (size_t)blockIdx.x*D_;
  for(int i = threadIdx.x; i < D_; i += 256)
    yr[i] = (_Float16)((xr[i]-mu)*rs*g[i] + be[i]);
}

// ---------------- small helper kernels ----------------
__global__ void cvt_k(const float* __restrict__ in, _Float16* __restrict__ out, size_t n){
  size_t i = (size_t)blockIdx.x*256 + threadIdx.x;
  if (i < n) out[i] = (_Float16)in[i];
}
// w[K][N] f32 -> wt[N][K] f16
__global__ void wtr_k(const float* __restrict__ w, _Float16* __restrict__ wt, int K, int N){
  size_t i = (size_t)blockIdx.x*256 + threadIdx.x;
  if (i < (size_t)K*N){
    int k = (int)(i / N), n = (int)(i % N);
    wt[(size_t)n*K + k] = (_Float16)w[i];
  }
}
// v[B,S,D] f16 -> vt[B,H,DH,S] f16
__global__ void vtr_k(const _Float16* __restrict__ v, _Float16* __restrict__ vt){
  size_t i = (size_t)blockIdx.x*256 + threadIdx.x;
  if (i < (size_t)M_*D_){
    int d  = (int)(i % D_);
    int bs = (int)(i / D_);
    int s = bs % S_, b = bs / S_;
    int h = d / DH_, dh = d % DH_;
    vt[((size_t)(b*H_ + h)*DH_ + dh)*S_ + s] = v[i];
  }
}

extern "C" void kernel_launch(void* const* d_in, const int* in_sizes, int n_in,
                              void* d_out, int out_size, void* d_ws, size_t ws_size,
                              hipStream_t stream){
  (void)in_sizes; (void)n_in; (void)out_size; (void)ws_size;
  const float* queries = (const float*)d_in[0];
  const float* memory_ = (const float*)d_in[1];
  // d_in[2] = causal mask (recomputed analytically in flash_k<true>)
  const float *ca_qw=(const float*)d_in[3],  *ca_qb=(const float*)d_in[4];
  const float *ca_kw=(const float*)d_in[5],  *ca_kb=(const float*)d_in[6];
  const float *ca_vw=(const float*)d_in[7],  *ca_vb=(const float*)d_in[8];
  const float *ca_ow=(const float*)d_in[9],  *ca_ob=(const float*)d_in[10];
  const float *sa_qw=(const float*)d_in[11], *sa_qb=(const float*)d_in[12];
  const float *sa_kw=(const float*)d_in[13], *sa_kb=(const float*)d_in[14];
  const float *sa_vw=(const float*)d_in[15], *sa_vb=(const float*)d_in[16];
  const float *sa_ow=(const float*)d_in[17], *sa_ob=(const float*)d_in[18];
  const float *w1=(const float*)d_in[19], *b1=(const float*)d_in[20];
  const float *w2=(const float*)d_in[21], *b2=(const float*)d_in[22];
  const float *ln1_g=(const float*)d_in[23], *ln1_b=(const float*)d_in[24];
  const float *ln2_g=(const float*)d_in[25], *ln2_b=(const float*)d_in[26];
  const float *ln3_g=(const float*)d_in[27], *ln3_b=(const float*)d_in[28];

  char* wsp = (char*)d_ws;
  auto alloc = [&](size_t bytes)->void*{ void* p = (void*)wsp; wsp += (bytes + 255) & ~(size_t)255; return p; };
  _Float16* wt_caq = (_Float16*)alloc((size_t)D_*D_*2);
  _Float16* wt_cak = (_Float16*)alloc((size_t)D_*D_*2);
  _Float16* wt_cav = (_Float16*)alloc((size_t)D_*D_*2);
  _Float16* wt_cao = (_Float16*)alloc((size_t)D_*D_*2);
  _Float16* wt_saq = (_Float16*)alloc((size_t)D_*D_*2);
  _Float16* wt_sak = (_Float16*)alloc((size_t)D_*D_*2);
  _Float16* wt_sav = (_Float16*)alloc((size_t)D_*D_*2);
  _Float16* wt_sao = (_Float16*)alloc((size_t)D_*D_*2);
  _Float16* w1t    = (_Float16*)alloc((size_t)D_*FF_*2);
  _Float16* w2t    = (_Float16*)alloc((size_t)FF_*D_*2);
  _Float16* xln    = (_Float16*)alloc((size_t)M_*D_*2);
  _Float16* membf  = (_Float16*)alloc((size_t)M_*D_*2);
  _Float16* qbf    = (_Float16*)alloc((size_t)M_*D_*2);
  _Float16* kbf    = (_Float16*)alloc((size_t)M_*D_*2);
  _Float16* vbf    = (_Float16*)alloc((size_t)M_*D_*2);
  _Float16* vtb    = (_Float16*)alloc((size_t)M_*D_*2);
  _Float16* aout   = (_Float16*)alloc((size_t)M_*D_*2);
  float*    x1     = (float*)alloc((size_t)M_*D_*4);
  float*    x2     = (float*)alloc((size_t)M_*D_*4);
  _Float16* hbuf   = (_Float16*)alloc((size_t)M_*FF_*2);

  dim3 blk(256);
  const int nDD = D_*D_/256, nDF = D_*FF_/256, nMD = M_*D_/256;
  wtr_k<<<nDD, blk, 0, stream>>>(ca_qw, wt_caq, D_, D_);
  wtr_k<<<nDD, blk, 0, stream>>>(ca_kw, wt_cak, D_, D_);
  wtr_k<<<nDD, blk, 0, stream>>>(ca_vw, wt_cav, D_, D_);
  wtr_k<<<nDD, blk, 0, stream>>>(ca_ow, wt_cao, D_, D_);
  wtr_k<<<nDD, blk, 0, stream>>>(sa_qw, wt_saq, D_, D_);
  wtr_k<<<nDD, blk, 0, stream>>>(sa_kw, wt_sak, D_, D_);
  wtr_k<<<nDD, blk, 0, stream>>>(sa_vw, wt_sav, D_, D_);
  wtr_k<<<nDD, blk, 0, stream>>>(sa_ow, wt_sao, D_, D_);
  wtr_k<<<nDF, blk, 0, stream>>>(w1, w1t, D_, FF_);
  wtr_k<<<nDF, blk, 0, stream>>>(w2, w2t, FF_, D_);
  cvt_k<<<nMD, blk, 0, stream>>>(memory_, membf, (size_t)M_*D_);

  dim3 gD(M_/128, D_/128);      // 32 x 8 blocks
  dim3 gF(M_/128, FF_/128);     // 32 x 32
  dim3 gA(S_/128, H_, B_);      // 16 x 16 x 2

  // ---- cross-attention block ----
  ln_k<<<M_, blk, 0, stream>>>(queries, ln1_g, ln1_b, xln);
  gemm_k<D_,D_,false,false,true ><<<gD, blk, 0, stream>>>(xln,   wt_caq, ca_qb, nullptr, qbf);
  gemm_k<D_,D_,false,false,true ><<<gD, blk, 0, stream>>>(membf, wt_cak, ca_kb, nullptr, kbf);
  gemm_k<D_,D_,false,false,true ><<<gD, blk, 0, stream>>>(membf, wt_cav, ca_vb, nullptr, vbf);
  vtr_k<<<nMD, blk, 0, stream>>>(vbf, vtb);
  flash_k<false><<<gA, blk, 0, stream>>>(qbf, kbf, vtb, aout);
  gemm_k<D_,D_,false,true ,false><<<gD, blk, 0, stream>>>(aout, wt_cao, ca_ob, queries, x1);

  // ---- causal self-attention block ----
  ln_k<<<M_, blk, 0, stream>>>(x1, ln2_g, ln2_b, xln);
  gemm_k<D_,D_,false,false,true ><<<gD, blk, 0, stream>>>(xln, wt_saq, sa_qb, nullptr, qbf);
  gemm_k<D_,D_,false,false,true ><<<gD, blk, 0, stream>>>(xln, wt_sak, sa_kb, nullptr, kbf);
  gemm_k<D_,D_,false,false,true ><<<gD, blk, 0, stream>>>(xln, wt_sav, sa_vb, nullptr, vbf);
  vtr_k<<<nMD, blk, 0, stream>>>(vbf, vtb);
  flash_k<true><<<gA, blk, 0, stream>>>(qbf, kbf, vtb, aout);
  gemm_k<D_,D_,false,true ,false><<<gD, blk, 0, stream>>>(aout, wt_sao, sa_ob, x1, x2);

  // ---- FFN block ----
  ln_k<<<M_, blk, 0, stream>>>(x2, ln3_g, ln3_b, xln);
  gemm_k<FF_,D_,true ,false,true ><<<gF, blk, 0, stream>>>(xln,  w1t, b1, nullptr, hbuf);
  gemm_k<D_,FF_,false,true ,false><<<gD, blk, 0, stream>>>(hbuf, w2t, b2, x2, d_out);
}
